// VidHRFormerBlockEnc_55911884259735
// MI455X (gfx1250) — compile-verified, matrix operand order checked by reading, not run on hardware
//
#include <hip/hip_runtime.h>
#include <hip/hip_bf16.h>
#include <math.h>

typedef __hip_bfloat16 bf16;
typedef __attribute__((ext_vector_type(16))) __bf16 v16bf;
typedef __attribute__((ext_vector_type(8)))  float  v8f;

#define M_TOK 50176      // N*T*H*W tokens
#define CDIM  256
#define HWSZ  3136       // 56*56
#define TT    8

struct alignas(16) B128 { unsigned int w[4]; };
union Frag16 { v16bf v; B128 q[2]; };

__device__ __forceinline__ float gelu_erf(float x) {
    return 0.5f * x * (1.0f + erff(x * 0.70710678118654752f));
}

// Load one 16x32 bf16 WMMA fragment (A layout; B uses same pattern with
// column n <- contiguous row of the row-major operand).
__device__ __forceinline__ v16bf load_frag_g(const bf16* __restrict__ base, int ld,
                                             int row, int k0, int lane) {
    int r  = row + (lane & 15);
    int kk = k0 + ((lane & 16) ? 8 : 0);
    const bf16* p = base + (size_t)r * (size_t)ld + kk;
    Frag16 f;
    f.q[0] = *(const B128*)(p);
    f.q[1] = *(const B128*)(p + 16);
    return f.v;
}

// ---------------------------------------------------------------------------
// Generic GEMM: out[m,n] = act( (sum_k A[m,k]*W[n,k]) * sc[n] + sh[n] )
// A: MxK bf16 row-major (M multiple of 128), W: NnxK bf16 row-major.
// resOut != null -> fp32 residual accumulate (ld = Nn); else bf16 store.
// ---------------------------------------------------------------------------
__global__ __launch_bounds__(256) void gemm_bf16_wmma(
    const bf16* __restrict__ A, const bf16* __restrict__ W,
    const float* __restrict__ sc, const float* __restrict__ sh,
    float* __restrict__ resOut, bf16* __restrict__ bfOut,
    int Nn, int K, int doGelu)
{
    int lane = threadIdx.x & 31;
    int wave = threadIdx.x >> 5;
    int wm = wave >> 1, wn = wave & 1;
    int m0 = blockIdx.x * 128 + wm * 32;
    int n0 = blockIdx.y * 128 + wn * 64;

    v8f acc[2][4] = {};
    for (int k0 = 0; k0 < K; k0 += 32) {
        v16bf a0 = load_frag_g(A, K, m0,      k0, lane);
        v16bf a1 = load_frag_g(A, K, m0 + 16, k0, lane);
#pragma unroll
        for (int j = 0; j < 4; ++j) {
            v16bf b = load_frag_g(W, K, n0 + j * 16, k0, lane);
            acc[0][j] = __builtin_amdgcn_wmma_f32_16x16x32_bf16(false, a0, false, b,
                                (short)0, acc[0][j], false, false);
            acc[1][j] = __builtin_amdgcn_wmma_f32_16x16x32_bf16(false, a1, false, b,
                                (short)0, acc[1][j], false, false);
        }
    }

    int hi8 = (lane & 16) ? 8 : 0;
    int cb  = lane & 15;
#pragma unroll
    for (int i = 0; i < 2; ++i)
#pragma unroll
        for (int j = 0; j < 4; ++j) {
            int colg = n0 + j * 16 + cb;
            float s = sc[colg], t = sh[colg];
#pragma unroll
            for (int v = 0; v < 8; ++v) {
                int row = m0 + i * 16 + hi8 + v;
                float y = acc[i][j][v] * s + t;
                if (doGelu) y = gelu_erf(y);
                if (resOut) resOut[(size_t)row * Nn + colg] += y;
                else        bfOut [(size_t)row * Nn + colg] = __float2bfloat16(y);
            }
        }
}

// ---------------------------------------------------------------------------
// Fused windowed spatial attention: one wave per (window, head).
// qk: Mx512 bf16 (q scaled+biased | k biased), vbuf: Mx256 bf16.
// ---------------------------------------------------------------------------
#define SA_WAVES 4
__global__ __launch_bounds__(SA_WAVES * 32) void spatial_attn_kernel(
    const bf16* __restrict__ qk, const bf16* __restrict__ vbuf, bf16* __restrict__ outb)
{
    __shared__ __align__(16) bf16 ldsVt[SA_WAVES][32 * 64];  // V^T, d x j
    __shared__ __align__(16) bf16 ldsP [SA_WAVES][64 * 64];  // softmax probs
    int lane = threadIdx.x & 31;
    int wave = threadIdx.x >> 5;
    int gw   = blockIdx.x * SA_WAVES + wave;   // 0..8191
    int win  = gw >> 3, head = gw & 7;
    int nt = win >> 6, qh = (win >> 3) & 7, qw = win & 7;
    int base_hw = nt * HWSZ;
    auto tok = [&](int p) {
        int ph = p / 7, pw = p - ph * 7;
        return base_hw + (qh * 7 + ph) * 56 + (qw * 7 + pw);
    };

    // stage V^T (pad j to 64 with zeros)
    for (int j = lane; j < 64; j += 32) {
        if (j < 49) {
            const bf16* vp = vbuf + (size_t)tok(j) * CDIM + head * 32;
#pragma unroll
            for (int dd = 0; dd < 32; ++dd) ldsVt[wave][dd * 64 + j] = vp[dd];
        } else {
#pragma unroll
            for (int dd = 0; dd < 32; ++dd) ldsVt[wave][dd * 64 + j] = __float2bfloat16(0.f);
        }
    }

    int hi8 = (lane & 16) ? 8 : 0;
    int cb  = lane & 15;
    int d0  = (lane & 16) ? 8 : 0;

    // S = Q K^T per 16-row block, softmax, P -> LDS
    for (int mt = 0; mt < 4; ++mt) {
        int iq = mt * 16 + cb; if (iq > 48) iq = 48;
        const bf16* qp = qk + (size_t)tok(iq) * 512 + head * 32 + d0;
        Frag16 fa; fa.q[0] = *(const B128*)qp; fa.q[1] = *(const B128*)(qp + 16);
        v8f s[4];
#pragma unroll
        for (int ntl = 0; ntl < 4; ++ntl) {
            int jk = ntl * 16 + cb; if (jk > 48) jk = 48;
            const bf16* kp = qk + (size_t)tok(jk) * 512 + 256 + head * 32 + d0;
            Frag16 fb; fb.q[0] = *(const B128*)kp; fb.q[1] = *(const B128*)(kp + 16);
            v8f z = {};
            s[ntl] = __builtin_amdgcn_wmma_f32_16x16x32_bf16(false, fa.v, false, fb.v,
                             (short)0, z, false, false);
        }
#pragma unroll
        for (int ntl = 0; ntl < 4; ++ntl)
            if (ntl * 16 + cb >= 49) {
#pragma unroll
                for (int v = 0; v < 8; ++v) s[ntl][v] = -1e30f;
            }
#pragma unroll
        for (int v = 0; v < 8; ++v) {
            float mx = fmaxf(fmaxf(s[0][v], s[1][v]), fmaxf(s[2][v], s[3][v]));
            for (int off = 8; off; off >>= 1) mx = fmaxf(mx, __shfl_xor(mx, off, 16));
            float e[4]; float sum = 0.f;
#pragma unroll
            for (int ntl = 0; ntl < 4; ++ntl) { e[ntl] = __expf(s[ntl][v] - mx); sum += e[ntl]; }
            for (int off = 8; off; off >>= 1) sum += __shfl_xor(sum, off, 16);
            float rinv = 1.f / sum;
            int row = mt * 16 + hi8 + v;
#pragma unroll
            for (int ntl = 0; ntl < 4; ++ntl)
                ldsP[wave][row * 64 + ntl * 16 + cb] = __float2bfloat16(e[ntl] * rinv);
        }
    }

    // O = P V
#pragma unroll
    for (int mt = 0; mt < 4; ++mt) {
#pragma unroll
        for (int vt = 0; vt < 2; ++vt) {
            v8f o = {};
#pragma unroll
            for (int kt = 0; kt < 2; ++kt) {
                int j0 = kt * 32 + d0;
                const bf16* pp = &ldsP [wave][(mt * 16 + cb) * 64 + j0];
                const bf16* vp = &ldsVt[wave][(vt * 16 + cb) * 64 + j0];
                Frag16 fp, fv;
                fp.q[0] = *(const B128*)pp; fp.q[1] = *(const B128*)(pp + 16);
                fv.q[0] = *(const B128*)vp; fv.q[1] = *(const B128*)(vp + 16);
                o = __builtin_amdgcn_wmma_f32_16x16x32_bf16(false, fp.v, false, fv.v,
                            (short)0, o, false, false);
            }
#pragma unroll
            for (int v = 0; v < 8; ++v) {
                int i = mt * 16 + hi8 + v;
                if (i < 49)
                    outb[(size_t)tok(i) * CDIM + head * 32 + vt * 16 + cb] = __float2bfloat16(o[v]);
            }
        }
    }
}

// ---------------------------------------------------------------------------
// Temporal attention, L=8 (tiny) -> scalar per (batch, head, i)
// ---------------------------------------------------------------------------
__global__ __launch_bounds__(256) void temporal_attn_kernel(
    const bf16* __restrict__ qk, const bf16* __restrict__ vbuf, bf16* __restrict__ outb)
{
    int id = blockIdx.x * 256 + threadIdx.x;     // 6272*8*8 = 401408 exactly
    int i = id & 7;
    int head = (id >> 3) & 7;
    int b = id >> 6;
    int n = b / HWSZ, hw = b - n * HWSZ;
    size_t tbase = (size_t)(n * TT) * HWSZ + hw;
    const bf16* qrow = qk + (tbase + (size_t)i * HWSZ) * 512 + head * 32;
    float s[8]; float mx = -1e30f;
    for (int j = 0; j < 8; ++j) {
        const bf16* krow = qk + (tbase + (size_t)j * HWSZ) * 512 + 256 + head * 32;
        float a = 0.f;
        for (int dd = 0; dd < 32; ++dd)
            a += __bfloat162float(qrow[dd]) * __bfloat162float(krow[dd]);
        s[j] = a; mx = fmaxf(mx, a);
    }
    float sum = 0.f;
    for (int j = 0; j < 8; ++j) { s[j] = __expf(s[j] - mx); sum += s[j]; }
    float rinv = 1.f / sum;
    bf16* orow = outb + (tbase + (size_t)i * HWSZ) * CDIM + head * 32;
    for (int dd = 0; dd < 32; ++dd) {
        float a = 0.f;
        for (int j = 0; j < 8; ++j)
            a += s[j] * __bfloat162float(vbuf[(tbase + (size_t)j * HWSZ) * CDIM + head * 32 + dd]);
        orow[dd] = __float2bfloat16(a * rinv);
    }
}

// ---------------------------------------------------------------------------
// LayerNorm over C=256 (one wave/token); emits bf16 xn and optional xn+pos.
// pos_mode: 0 none, 1 spatial window pos, 2 temporal pos
// ---------------------------------------------------------------------------
__global__ __launch_bounds__(256) void ln_kernel(
    const float* __restrict__ x, const float* __restrict__ g, const float* __restrict__ b,
    bf16* __restrict__ xn, bf16* __restrict__ qkout, const float* __restrict__ pos, int pos_mode)
{
    int lane = threadIdx.x & 31;
    int wave = threadIdx.x >> 5;
    size_t tokı = (size_t)blockIdx.x * 8 + wave;
    const float4* r4 = (const float4*)(x + tokı * CDIM);
    float4 a = r4[lane * 2], c = r4[lane * 2 + 1];
    float vs[8] = {a.x, a.y, a.z, a.w, c.x, c.y, c.z, c.w};
    float s = 0.f, ss = 0.f;
#pragma unroll
    for (int e = 0; e < 8; ++e) { s += vs[e]; ss += vs[e] * vs[e]; }
    for (int off = 16; off; off >>= 1) { s += __shfl_xor(s, off, 32); ss += __shfl_xor(ss, off, 32); }
    float mean = s * (1.f / 256.f);
    float var  = ss * (1.f / 256.f) - mean * mean;
    float inv  = rsqrtf(var + 1e-5f);
    int pidx = 0;
    if (pos_mode == 1) {
        int hh = (int)((tokı / 56) % 56); int ww = (int)(tokı % 56);
        pidx = ((hh % 7) * 7 + (ww % 7)) * CDIM;
    } else if (pos_mode == 2) {
        pidx = (int)((tokı / HWSZ) % TT) * CDIM;
    }
    B128 px, pq; bf16* phx = (bf16*)&px; bf16* phq = (bf16*)&pq;
    int col0 = lane * 8;
#pragma unroll
    for (int e = 0; e < 8; ++e) {
        int col = col0 + e;
        float y = (vs[e] - mean) * inv * g[col] + b[col];
        phx[e] = __float2bfloat16(y);
        if (qkout) phq[e] = __float2bfloat16(y + pos[pidx + col]);
    }
    ((B128*)(xn + tokı * CDIM))[lane] = px;
    if (qkout) ((B128*)(qkout + tokı * CDIM))[lane] = pq;
}

// depthwise 3x3 + BN2 + GELU (bn/bias folded into sc/sh)
__global__ __launch_bounds__(256) void dwconv_kernel(
    const bf16* __restrict__ h1, const float* __restrict__ dww,
    const float* __restrict__ sc, const float* __restrict__ sh, bf16* __restrict__ h2)
{
    int bidx = blockIdx.x;                 // 50176 spatial positions
    int ntc = bidx / HWSZ, rem = bidx - ntc * HWSZ;
    int hh = rem / 56, ww = rem - hh * 56;
    size_t base = (size_t)bidx * 1024;
    for (int c = threadIdx.x; c < 1024; c += 256) {
        float acc = 0.f;
#pragma unroll
        for (int dy = 0; dy < 3; ++dy) {
            int y = hh + dy - 1; if (y < 0 || y >= 56) continue;
#pragma unroll
            for (int dx = 0; dx < 3; ++dx) {
                int xx = ww + dx - 1; if (xx < 0 || xx >= 56) continue;
                size_t off = ((size_t)ntc * HWSZ + y * 56 + xx) * 1024 + c;
                acc += __bfloat162float(h1[off]) * dww[c * 9 + dy * 3 + dx];
            }
        }
        h2[base + c] = __float2bfloat16(gelu_erf(acc * sc[c] + sh[c]));
    }
}

__global__ void build_affine_kernel(float* sc, float* sh, const float* bias,
    const float* g, const float* b, const float* m, const float* v,
    int n, float qs, int qcount)
{
    int i = blockIdx.x * 256 + threadIdx.x;
    if (i >= n) return;
    float s = 1.f, t = bias ? bias[i] : 0.f;
    if (g) { float inv = rsqrtf(v[i] + 1e-5f); s = g[i] * inv; t = (t - m[i]) * s + b[i]; }
    if (i < qcount) { s *= qs; t *= qs; }
    sc[i] = s; sh[i] = t;
}

__global__ void f32_to_bf16_kernel(const float* __restrict__ in, bf16* __restrict__ out, int n) {
    int i = blockIdx.x * 256 + threadIdx.x;
    if (i < n) out[i] = __float2bfloat16(in[i]);
}

__global__ void copy_f32_kernel(const float* __restrict__ in, float* __restrict__ out) {
    size_t i = ((size_t)blockIdx.x * 256 + threadIdx.x) * 4;
    *(float4*)(out + i) = *(const float4*)(in + i);
}

// ---------------------------------------------------------------------------
extern "C" void kernel_launch(void* const* d_in, const int* in_sizes, int n_in,
                              void* d_out, int out_size, void* d_ws, size_t ws_size,
                              hipStream_t stream)
{
    (void)in_sizes; (void)n_in; (void)out_size; (void)ws_size;
    const float* x            = (const float*)d_in[0];
    const float* local_pos    = (const float*)d_in[1];
    const float* temporal_pos = (const float*)d_in[2];
    const float* ln1_g = (const float*)d_in[3],  *ln1_b = (const float*)d_in[4];
    const float* ln2_g = (const float*)d_in[5],  *ln2_b = (const float*)d_in[6];
    const float* ln3_g = (const float*)d_in[7],  *ln3_b = (const float*)d_in[8];
    const float* ln4_g = (const float*)d_in[9],  *ln4_b = (const float*)d_in[10];
    const float* s_in_w  = (const float*)d_in[11], *s_in_b  = (const float*)d_in[12];
    const float* s_out_w = (const float*)d_in[13], *s_out_b = (const float*)d_in[14];
    const float* t_in_w  = (const float*)d_in[15], *t_in_b  = (const float*)d_in[16];
    const float* t_out_w = (const float*)d_in[17], *t_out_b = (const float*)d_in[18];
    const float* fc1_w = (const float*)d_in[19], *fc1_b = (const float*)d_in[20];
    const float* dw_w  = (const float*)d_in[21], *dw_b  = (const float*)d_in[22];
    const float* fc2_w = (const float*)d_in[23], *fc2_b = (const float*)d_in[24];
    const float* bn1_g = (const float*)d_in[25], *bn1_b = (const float*)d_in[26];
    const float* bn1_m = (const float*)d_in[27], *bn1_v = (const float*)d_in[28];
    const float* bn2_g = (const float*)d_in[29], *bn2_b = (const float*)d_in[30];
    const float* bn2_m = (const float*)d_in[31], *bn2_v = (const float*)d_in[32];
    const float* bn3_g = (const float*)d_in[33], *bn3_b = (const float*)d_in[34];
    const float* bn3_m = (const float*)d_in[35], *bn3_v = (const float*)d_in[36];
    const float* lin1_w = (const float*)d_in[37], *lin1_b = (const float*)d_in[38];
    const float* lin2_w = (const float*)d_in[39], *lin2_b = (const float*)d_in[40];
    float* res = (float*)d_out;   // residual accumulator == final output

    char* ws = (char*)d_ws;
    size_t off = 0;
    auto carve = [&](size_t bytes) { void* p = ws + off; off += (bytes + 255) & ~(size_t)255; return p; };
    bf16* wb_sin  = (bf16*)carve((size_t)768  * 256 * 2);
    bf16* wb_sout = (bf16*)carve((size_t)256  * 256 * 2);
    bf16* wb_tin  = (bf16*)carve((size_t)768  * 256 * 2);
    bf16* wb_tout = (bf16*)carve((size_t)256  * 256 * 2);
    bf16* wb_fc1  = (bf16*)carve((size_t)1024 * 256 * 2);
    bf16* wb_fc2  = (bf16*)carve((size_t)256  * 1024 * 2);
    bf16* wb_lin1 = (bf16*)carve((size_t)1024 * 256 * 2);
    bf16* wb_lin2 = (bf16*)carve((size_t)256  * 1024 * 2);
    float* aff = (float*)carve(16384 * 4);
    float *sc_sqkv = aff + 0,     *sh_sqkv = aff + 768;
    float *sc_sout = aff + 1536,  *sh_sout = aff + 1792;
    float *sc_fc1  = aff + 2048,  *sh_fc1  = aff + 3072;
    float *sc_dw   = aff + 4096,  *sh_dw   = aff + 5120;
    float *sc_fc2  = aff + 6144,  *sh_fc2  = aff + 6400;
    float *sc_tqkv = aff + 6656,  *sh_tqkv = aff + 7424;
    float *sc_tout = aff + 8192,  *sh_tout = aff + 8448;
    float *sc_l1   = aff + 8704,  *sh_l1   = aff + 9728;
    float *sc_l2   = aff + 10752, *sh_l2   = aff + 11008;
    bf16* xn   = (bf16*)carve((size_t)M_TOK * 256 * 2);
    bf16* qkb  = (bf16*)carve((size_t)M_TOK * 256 * 2);
    bf16* big1 = (bf16*)carve((size_t)M_TOK * 1024 * 2);  // qko|vo|attn, later h1
    bf16* big2 = (bf16*)carve((size_t)M_TOK * 1024 * 2);  // h2, later FFN hidden
    bf16* qko  = big1;
    bf16* vo   = big1 + (size_t)M_TOK * 512;
    bf16* attn = big1 + (size_t)M_TOK * 768;

    const float rs = 0.17677669529663687f;  // 1/sqrt(32)

    copy_f32_kernel<<<12544, 256, 0, stream>>>(x, res);

    auto cvt = [&](const float* w, bf16* o, int n) {
        f32_to_bf16_kernel<<<(n + 255) / 256, 256, 0, stream>>>(w, o, n);
    };
    cvt(s_in_w, wb_sin, 768 * 256);   cvt(s_out_w, wb_sout, 256 * 256);
    cvt(t_in_w, wb_tin, 768 * 256);   cvt(t_out_w, wb_tout, 256 * 256);
    cvt(fc1_w, wb_fc1, 1024 * 256);   cvt(fc2_w, wb_fc2, 256 * 1024);
    cvt(lin1_w, wb_lin1, 1024 * 256); cvt(lin2_w, wb_lin2, 256 * 1024);

    auto ba = [&](float* sc, float* sh, const float* bias, const float* g, const float* b,
                  const float* m, const float* v, int n, float qs, int qc) {
        build_affine_kernel<<<(n + 255) / 256, 256, 0, stream>>>(sc, sh, bias, g, b, m, v, n, qs, qc);
    };
    ba(sc_sqkv, sh_sqkv, s_in_b, nullptr, nullptr, nullptr, nullptr, 768, rs, 256);
    ba(sc_sout, sh_sout, s_out_b, nullptr, nullptr, nullptr, nullptr, 256, 1.f, 0);
    ba(sc_fc1,  sh_fc1,  fc1_b, bn1_g, bn1_b, bn1_m, bn1_v, 1024, 1.f, 0);
    ba(sc_dw,   sh_dw,   dw_b,  bn2_g, bn2_b, bn2_m, bn2_v, 1024, 1.f, 0);
    ba(sc_fc2,  sh_fc2,  fc2_b, bn3_g, bn3_b, bn3_m, bn3_v, 256, 1.f, 0);
    ba(sc_tqkv, sh_tqkv, t_in_b, nullptr, nullptr, nullptr, nullptr, 768, rs, 256);
    ba(sc_tout, sh_tout, t_out_b, nullptr, nullptr, nullptr, nullptr, 256, 1.f, 0);
    ba(sc_l1,   sh_l1,   lin1_b, nullptr, nullptr, nullptr, nullptr, 1024, 1.f, 0);
    ba(sc_l2,   sh_l2,   lin2_b, nullptr, nullptr, nullptr, nullptr, 256, 1.f, 0);

    auto gemm = [&](const bf16* A, const bf16* W, const float* sc, const float* sh,
                    float* ro, bf16* bo, int Nn, int K, int gl) {
        dim3 grid(M_TOK / 128, Nn / 128);
        gemm_bf16_wmma<<<grid, dim3(256), 0, stream>>>(A, W, sc, sh, ro, bo, Nn, K, gl);
    };

    // 1) windowed spatial attention
    ln_kernel<<<M_TOK / 8, 256, 0, stream>>>(res, ln1_g, ln1_b, xn, qkb, local_pos, 1);
    gemm(qkb, wb_sin,             sc_sqkv,       sh_sqkv,       nullptr, qko, 512, 256, 0);
    gemm(xn,  wb_sin + 512 * 256, sc_sqkv + 512, sh_sqkv + 512, nullptr, vo,  256, 256, 0);
    spatial_attn_kernel<<<2048, SA_WAVES * 32, 0, stream>>>(qko, vo, attn);
    gemm(attn, wb_sout, sc_sout, sh_sout, res, nullptr, 256, 256, 0);

    // 2) MlpDWBN spatial FFN
    ln_kernel<<<M_TOK / 8, 256, 0, stream>>>(res, ln2_g, ln2_b, xn, nullptr, nullptr, 0);
    gemm(xn, wb_fc1, sc_fc1, sh_fc1, nullptr, big1, 1024, 256, 1);
    dwconv_kernel<<<M_TOK, 256, 0, stream>>>(big1, dw_w, sc_dw, sh_dw, big2);
    gemm(big2, wb_fc2, sc_fc2, sh_fc2, res, nullptr, 256, 1024, 1);

    // 3) temporal attention
    ln_kernel<<<M_TOK / 8, 256, 0, stream>>>(res, ln3_g, ln3_b, xn, qkb, temporal_pos, 2);
    gemm(qkb, wb_tin,             sc_tqkv,       sh_tqkv,       nullptr, qko, 512, 256, 0);
    gemm(xn,  wb_tin + 512 * 256, sc_tqkv + 512, sh_tqkv + 512, nullptr, vo,  256, 256, 0);
    temporal_attn_kernel<<<1568, 256, 0, stream>>>(qko, vo, attn);
    gemm(attn, wb_tout, sc_tout, sh_tout, res, nullptr, 256, 256, 0);

    // 4) temporal FFN
    ln_kernel<<<M_TOK / 8, 256, 0, stream>>>(res, ln4_g, ln4_b, xn, nullptr, nullptr, 0);
    gemm(xn, wb_lin1, sc_l1, sh_l1, nullptr, big2, 1024, 256, 1);
    gemm(big2, wb_lin2, sc_l2, sh_l2, res, nullptr, 256, 1024, 0);
}
